// BigramLanguageModel_3934190043171
// MI455X (gfx1250) — matrix operand order repaired and checked by reading, not compiled, not run on hardware
//
#include <hip/hip_runtime.h>
#include <math.h>

#define DEVINL __device__ __forceinline__

typedef __bf16 bf16_t;
typedef bf16_t v16bf __attribute__((ext_vector_type(16)));
typedef bf16_t v8bf  __attribute__((ext_vector_type(8)));
typedef float  v8f   __attribute__((ext_vector_type(8)));

// ---- model dims (fixed by the reference) ----
constexpr int V_ = 8192, C_ = 512, H_ = 16, L_ = 12, HS_ = 32, FF_ = 2048, B_ = 8, T_ = 1024;
constexpr int M_ = B_ * T_;   // 8192 tokens

DEVINL v8f wmma_bf16(v16bf a, v16bf b, v8f c) {
  // D = A(16x32 bf16) * B(32x16 bf16) + C(16x16 f32)
  return __builtin_amdgcn_wmma_f32_16x16x32_bf16(false, a, false, b, (short)0, c, false, false);
}

DEVINL v16bf pack16(v8bf lo, v8bf hi) {
  return __builtin_shufflevector(lo, hi, 0,1,2,3,4,5,6,7,8,9,10,11,12,13,14,15);
}

// A-operand fragment (M=lane&15; K interleave: e<8 -> K=e+8*kh, e>=8 -> K=8+e+8*kh)
DEVINL v16bf load_a_bf16(const bf16_t* base, int ld) {
  int lane = threadIdx.x & 31;
  const bf16_t* p = base + (size_t)(lane & 15) * ld + 8 * (lane >> 4);
  v8bf lo = *(const v8bf*)p;
  v8bf hi = *(const v8bf*)(p + 16);
  return pack16(lo, hi);
}

// B-operand fragment: lane&15 = column n; K interleave: e<8 -> K=e+8*kh, e>=8 -> K=8+e+8*kh
DEVINL v16bf load_b_bf16(const bf16_t* base, int ld) {
  int lane = threadIdx.x & 31;
  const bf16_t* p = base + (size_t)(lane & 15) * ld + 8 * (lane >> 4);
  v8bf lo = *(const v8bf*)p;
  v8bf hi = *(const v8bf*)(p + 16);
  return pack16(lo, hi);
}

// ---------------- GEMM: out[M,N] = A[M,K](bf16) * Wt[N,K](bf16)^T + bias ----------------------------
// 4 waves per block stacked in M share one 64-wide weight tile.
// Software pipeline: B tiles double-buffered in LDS via global_load_async_to_lds_b128 (tiles i and
// i+1 always in flight); A fragments for iteration i+1 are register-prefetched during iteration i.
template<bool RELU, bool RESID, bool STORE_VT, bool OUT_BF16>
__global__ __launch_bounds__(128) void gemm_bf16_kernel(
    const bf16_t* __restrict__ A, const bf16_t* __restrict__ Wt,
    const float* __restrict__ bias, const float* __restrict__ resid,
    void* __restrict__ outv, int K, int ldout)
{
  __shared__ bf16_t bTile[2 * 64 * 40];      // double-buffered 64 n-rows x 32 k, stride 40 (80B)
  const int tid  = threadIdx.x;
  const int wave = tid >> 5;
  const int nBlk = blockIdx.x * 64;
  const int mBlk = blockIdx.y * 256 + wave * 64;

  v8f vz = {0.f,0.f,0.f,0.f,0.f,0.f,0.f,0.f};
  v8f acc[4][4];
  #pragma unroll
  for (int i = 0; i < 4; ++i)
    #pragma unroll
    for (int j = 0; j < 4; ++j) acc[i][j] = vz;

  const unsigned ldsBase = (unsigned)(uintptr_t)(void*)&bTile[0];  // LDS byte offset = low 32 bits

  // issue the async copy of weight tile `it` (64x32 bf16 = 4KB) into LDS buffer it&1
  auto issue_tile = [&](int it) {
    const int k0 = it << 5;
    const unsigned bufBase = ldsBase + (unsigned)(it & 1) * (64 * 80);
    #pragma unroll
    for (int j = 0; j < 2; ++j) {             // 256 x 16B chunks, 2 per thread
      int i = tid + 128 * j;
      int n = i >> 2, c = i & 3;
      unsigned long long ga =
          (unsigned long long)(uintptr_t)(Wt + (size_t)(nBlk + n) * K + k0 + c * 8);
      unsigned la = bufBase + n * 80 + c * 16;
      asm volatile("global_load_async_to_lds_b128 %0, %1, off"
                   :: "v"(la), "v"(ga) : "memory");
    }
  };

  const int nt = K >> 5;                      // number of 32-wide K tiles (>= 16, even)
  issue_tile(0);
  if (nt > 1) issue_tile(1);

  // register-prefetch A fragments for iteration 0
  v16bf af[4];
  #pragma unroll
  for (int fm = 0; fm < 4; ++fm)
    af[fm] = load_a_bf16(A + (size_t)(mBlk + 16 * fm) * K, K);

  for (int it = 0; it < nt; ++it) {
    // async loads complete in order; 2 instructions per tile per wave, tile it+1 is the newest
    if (it + 1 < nt) asm volatile("s_wait_asynccnt 2" ::: "memory");
    else             asm volatile("s_wait_asynccnt 0" ::: "memory");
    __syncthreads();                          // tile `it` visible to all 4 waves

    const bf16_t* buf = bTile + (it & 1) * (64 * 40);
    v16bf bfr[4];
    #pragma unroll
    for (int fn = 0; fn < 4; ++fn)
      bfr[fn] = load_b_bf16(buf + 16 * fn * 40, 40);

    #pragma unroll
    for (int fm = 0; fm < 4; ++fm)
      #pragma unroll
      for (int fn = 0; fn < 4; ++fn)
        acc[fm][fn] = wmma_bf16(af[fm], bfr[fn], acc[fm][fn]);

    // prefetch A fragments for the next iteration (latency hidden behind barrier + next ds reads)
    if (it + 1 < nt) {
      const int k1 = (it + 1) << 5;
      #pragma unroll
      for (int fm = 0; fm < 4; ++fm)
        af[fm] = load_a_bf16(A + (size_t)(mBlk + 16 * fm) * K + k1, K);
    }

    __syncthreads();                          // all waves done reading buffer it&1
    if (it + 2 < nt) issue_tile(it + 2);      // refill the buffer just drained
  }

  const int lane = threadIdx.x & 31;
  const int nl = lane & 15, mh = lane >> 4;
  #pragma unroll
  for (int fm = 0; fm < 4; ++fm)
    #pragma unroll
    for (int fn = 0; fn < 4; ++fn)
      #pragma unroll
      for (int r = 0; r < 8; ++r) {
        int m = mBlk + 16 * fm + r + 8 * mh;
        int n = nBlk + 16 * fn + nl;
        float v = acc[fm][fn][r];
        if (bias) v += bias[n];
        if (RELU) v = fmaxf(v, 0.f);
        size_t oi;
        if (STORE_VT) {  // write V transposed: vT[b][h][d][s]
          int b = m >> 10, s = m & (T_ - 1);
          int h = n >> 5,  d = n & (HS_ - 1);
          oi = (size_t)((b * H_ + h) * HS_ + d) * T_ + s;
        } else {
          oi = (size_t)m * ldout + n;
        }
        if (OUT_BF16) {
          ((bf16_t*)outv)[oi] = (bf16_t)v;
        } else {
          if (RESID) v += resid[oi];
          ((float*)outv)[oi] = v;
        }
      }
}

// ---------------- Flash attention: one wave handles 16 query rows of one (b,h) -----------------------
__global__ __launch_bounds__(32) void attn_kernel(
    const bf16_t* __restrict__ qk, const bf16_t* __restrict__ vT, bf16_t* __restrict__ o)
{
  __shared__ bf16_t pTile[16 * 40];           // 16 t-rows x 32 s-cols, padded stride 40
  const int tBlk = blockIdx.x * 16;
  const int bh = blockIdx.y;
  const int b = bh >> 4, h = bh & (H_ - 1);
  const int lane = threadIdx.x & 31;
  const int nl = lane & 15, mh = lane >> 4;
  const float scale = 0.17677669529663687f;   // HS^-0.5
  const float NEG_INF = -__builtin_inff();

  const int ldqk = 2 * C_;
  v16bf qf = load_a_bf16(qk + (size_t)(b * T_ + tBlk) * ldqk + h * HS_, ldqk);

  v8f vz = {0.f,0.f,0.f,0.f,0.f,0.f,0.f,0.f};
  v8f oacc0 = vz, oacc1 = vz;
  float mrow[8], lrow[8];
  #pragma unroll
  for (int r = 0; r < 8; ++r) { mrow[r] = NEG_INF; lrow[r] = 0.f; }

  const int limit = tBlk + 16;                // causal: s <= t < limit
  for (int s0 = 0; s0 < limit; s0 += 32) {
    v8f S0, S1 = vz;
    S0 = wmma_bf16(qf,
        load_b_bf16(qk + (size_t)(b * T_ + s0) * ldqk + C_ + h * HS_, ldqk), vz);
    const bool have1 = (s0 + 16 < limit);
    if (have1)
      S1 = wmma_bf16(qf,
          load_b_bf16(qk + (size_t)(b * T_ + s0 + 16) * ldqk + C_ + h * HS_, ldqk), vz);
    #pragma unroll
    for (int r = 0; r < 8; ++r) {
      int t = tBlk + r + 8 * mh;
      float v0 = (s0 + nl            <= t)      ? S0[r] * scale : NEG_INF;
      float v1 = (have1 && (s0 + 16 + nl <= t)) ? S1[r] * scale : NEG_INF;
      float mx = fmaxf(v0, v1);
      mx = fmaxf(mx, __shfl_xor(mx, 1));
      mx = fmaxf(mx, __shfl_xor(mx, 2));
      mx = fmaxf(mx, __shfl_xor(mx, 4));
      mx = fmaxf(mx, __shfl_xor(mx, 8));      // row max across the 16-lane half
      float mnew  = fmaxf(mrow[r], mx);
      float alpha = (mrow[r] == NEG_INF) ? 0.f : __expf(mrow[r] - mnew);
      float p0 = __expf(v0 - mnew);
      float p1 = __expf(v1 - mnew);
      float ps = p0 + p1;
      ps += __shfl_xor(ps, 1);
      ps += __shfl_xor(ps, 2);
      ps += __shfl_xor(ps, 4);
      ps += __shfl_xor(ps, 8);
      lrow[r] = lrow[r] * alpha + ps;
      mrow[r] = mnew;
      oacc0[r] *= alpha;
      oacc1[r] *= alpha;
      pTile[(r + 8 * mh) * 40 + nl]      = (bf16_t)p0;   // D-layout -> row-major via LDS
      pTile[(r + 8 * mh) * 40 + 16 + nl] = (bf16_t)p1;
    }
    // single wave: DS ops are in-order, no barrier needed
    v16bf pf = load_a_bf16(pTile, 40);
    oacc0 = wmma_bf16(pf,
        load_b_bf16(vT + (size_t)((b * H_ + h) * HS_ + 0) * T_ + s0, T_), oacc0);
    oacc1 = wmma_bf16(pf,
        load_b_bf16(vT + (size_t)((b * H_ + h) * HS_ + 16) * T_ + s0, T_), oacc1);
  }
  #pragma unroll
  for (int r = 0; r < 8; ++r) {
    int t = tBlk + r + 8 * mh;
    float inv = 1.0f / lrow[r];
    o[(size_t)(b * T_ + t) * C_ + h * HS_ +      nl] = (bf16_t)(oacc0[r] * inv);
    o[(size_t)(b * T_ + t) * C_ + h * HS_ + 16 + nl] = (bf16_t)(oacc1[r] * inv);
  }
}

// ---------------- LayerNorm: one wave per token row; emits bf16 for the matrix units ----------------
__global__ __launch_bounds__(32) void ln_kernel(const float* __restrict__ x,
    const float* __restrict__ g, const float* __restrict__ bta, bf16_t* __restrict__ out)
{
  const int row = blockIdx.x;
  const int lane = threadIdx.x;
  const float* p = x + (size_t)row * C_ + lane * 16;
  float v[16];
  #pragma unroll
  for (int i = 0; i < 16; ++i) v[i] = p[i];
  float s = 0.f, ss = 0.f;
  #pragma unroll
  for (int i = 0; i < 16; ++i) { s += v[i]; ss += v[i] * v[i]; }
  #pragma unroll
  for (int off = 1; off < 32; off <<= 1) { s += __shfl_xor(s, off); ss += __shfl_xor(ss, off); }
  float mean = s * (1.f / C_);
  float var  = ss * (1.f / C_) - mean * mean;
  float rstd = rsqrtf(var + 1e-5f);
  bf16_t* q = out + (size_t)row * C_ + lane * 16;
  #pragma unroll
  for (int i = 0; i < 16; ++i) {
    int c = lane * 16 + i;
    q[i] = (bf16_t)((v[i] - mean) * rstd * g[c] + bta[c]);
  }
}

// ---------------- Embedding gather ------------------------------------------------------------------
__global__ __launch_bounds__(128) void embed_kernel(const int* __restrict__ idx,
    const int* __restrict__ labs, const float* __restrict__ tok,
    const float* __restrict__ pos, const float* __restrict__ lab, float* __restrict__ x)
{
  const int m = blockIdx.x;
  const int c = threadIdx.x * 4;
  const int t = m & (T_ - 1);
  const int id = idx[m], lb = labs[m];
  const float4 a  = *(const float4*)(tok + (size_t)id * C_ + c);
  const float4 e  = *(const float4*)(lab + (size_t)lb * C_ + c);
  const float4 pp = *(const float4*)(pos + (size_t)t  * C_ + c);
  float4 r;
  r.x = a.x + e.x + pp.x; r.y = a.y + e.y + pp.y;
  r.z = a.z + e.z + pp.z; r.w = a.w + e.w + pp.w;
  *(float4*)(x + (size_t)m * C_ + c) = r;
}

// ---------------- Weight repack to bf16 [N][K] ------------------------------------------------------
__global__ __launch_bounds__(256) void repack_qkv_kernel(const float* __restrict__ Wq,
    const float* __restrict__ Wk, const float* __restrict__ Wv, bf16_t* __restrict__ dst)
{
  size_t i = (size_t)blockIdx.x * 256 + threadIdx.x;   // over L*1536*C
  int k = (int)(i & (C_ - 1));
  size_t r = i >> 9;
  int n = (int)(r % 1536);
  int l = (int)(r / 1536);
  int which = n >> 9;              // 0=q, 1=k, 2=v
  int nn = n & 511;
  int h = nn >> 5, d = nn & 31;
  const float* W = (which == 0) ? Wq : (which == 1) ? Wk : Wv;
  dst[i] = (bf16_t)W[(((size_t)l * H_ + h) * C_ + k) * HS_ + d];
}

__global__ __launch_bounds__(256) void repack_t_kernel(const float* __restrict__ src,
    bf16_t* __restrict__ dst, int N, int K)
{
  size_t i = (size_t)blockIdx.x * 256 + threadIdx.x;   // over N*K
  int k = (int)(i % K);
  int n = (int)(i / K);
  dst[i] = (bf16_t)src[(size_t)k * N + n];             // src is [K,N] row-major
}

// ---------------- Loss (deterministic fixed-order reductions) ---------------------------------------
__global__ __launch_bounds__(256) void loss_rows_kernel(const float* __restrict__ logits,
    const int* __restrict__ targets, float* __restrict__ terms)
{
  __shared__ float red[256];
  const int row = blockIdx.x, tid = threadIdx.x;
  const float* p = logits + (size_t)row * V_;
  float mx = -__builtin_inff();
  for (int i = tid; i < V_; i += 256) mx = fmaxf(mx, p[i]);
  red[tid] = mx; __syncthreads();
  for (int s = 128; s > 0; s >>= 1) { if (tid < s) red[tid] = fmaxf(red[tid], red[tid + s]); __syncthreads(); }
  mx = red[0]; __syncthreads();
  float sum = 0.f;
  for (int i = tid; i < V_; i += 256) sum += __expf(p[i] - mx);
  red[tid] = sum; __syncthreads();
  for (int s = 128; s > 0; s >>= 1) { if (tid < s) red[tid] += red[tid + s]; __syncthreads(); }
  if (tid == 0) terms[row] = mx + logf(red[0]) - p[targets[row]];
}

__global__ __launch_bounds__(256) void loss_reduce_kernel(const float* __restrict__ terms,
                                                          float* __restrict__ out)
{
  __shared__ float red[256];
  const int tid = threadIdx.x;
  float s = 0.f;
  for (int i = tid; i < M_; i += 256) s += terms[i];
  red[tid] = s; __syncthreads();
  for (int k = 128; k > 0; k >>= 1) { if (tid < k) red[tid] += red[tid + k]; __syncthreads(); }
  if (tid == 0) out[0] = red[0] * (1.0f / M_);
}

// ---------------- Host orchestration ----------------------------------------------------------------
extern "C" void kernel_launch(void* const* d_in, const int* in_sizes, int n_in,
                              void* d_out, int out_size, void* d_ws, size_t ws_size,
                              hipStream_t stream)
{
  (void)in_sizes; (void)n_in; (void)out_size; (void)ws_size;
  const int*   idx  = (const int*)d_in[0];
  const int*   labs = (const int*)d_in[1];
  const int*   tgt  = (const int*)d_in[2];
  const float* tok  = (const float*)d_in[3];
  const float* pos  = (const float*)d_in[4];
  const float* labe = (const float*)d_in[5];
  const float* Wq   = (const float*)d_in[6];
  const float* Wk   = (const float*)d_in[7];
  const float* Wv   = (const float*)d_in[8];
  const float* Wo   = (const float*)d_in[9];
  const float* bo   = (const float*)d_in[10];
  const float* ln1g = (const float*)d_in[11];
  const float* ln1b = (const float*)d_in[12];
  const float* W1   = (const float*)d_in[13];
  const float* b1   = (const float*)d_in[14];
  const float* W2   = (const float*)d_in[15];
  const float* b2   = (const float*)d_in[16];
  const float* ln2g = (const float*)d_in[17];
  const float* ln2b = (const float*)d_in[18];
  const float* lnfg = (const float*)d_in[19];
  const float* lnfb = (const float*)d_in[20];
  const float* Wlm  = (const float*)d_in[21];
  const float* blm  = (const float*)d_in[22];

  char* ws = (char*)d_ws;
  size_t off = 0;
  auto take = [&](size_t bytes) { void* p = ws + off; off += (bytes + 255) & ~(size_t)255; return p; };
  bf16_t* wqkvT = (bf16_t*)take((size_t)L_ * 1536 * C_ * 2);
  bf16_t* woT   = (bf16_t*)take((size_t)L_ * C_ * C_ * 2);
  bf16_t* w1T   = (bf16_t*)take((size_t)L_ * FF_ * C_ * 2);
  bf16_t* w2T   = (bf16_t*)take((size_t)L_ * C_ * FF_ * 2);
  bf16_t* wlmT  = (bf16_t*)take((size_t)V_ * C_ * 2);
  float*  X     = (float*) take((size_t)M_ * C_ * 4);   // residual stream stays f32
  bf16_t* Hb    = (bf16_t*)take((size_t)M_ * C_ * 2);   // LN output (GEMM A operand)
  bf16_t* QKb   = (bf16_t*)take((size_t)M_ * 2 * C_ * 2);
  bf16_t* VTb   = (bf16_t*)take((size_t)M_ * C_ * 2);
  bf16_t* Ob    = (bf16_t*)take((size_t)M_ * C_ * 2);
  bf16_t* FFb   = (bf16_t*)take((size_t)M_ * FF_ * 2);
  float*  terms = (float*) take((size_t)M_ * 4);

  // repack fp32 weights -> bf16 [N][K] (all bf16 weights ~85MB: resident in 192MB L2)
  repack_qkv_kernel<<<(L_ * 1536 * C_) / 256, 256, 0, stream>>>(Wq, Wk, Wv, wqkvT);
  for (int l = 0; l < L_; ++l) {
    repack_t_kernel<<<(C_ * C_) / 256, 256, 0, stream>>>(Wo + (size_t)l * C_ * C_,  woT + (size_t)l * C_ * C_,  C_,  C_);
    repack_t_kernel<<<(FF_ * C_) / 256, 256, 0, stream>>>(W1 + (size_t)l * C_ * FF_, w1T + (size_t)l * FF_ * C_, FF_, C_);
    repack_t_kernel<<<(FF_ * C_) / 256, 256, 0, stream>>>(W2 + (size_t)l * FF_ * C_, w2T + (size_t)l * C_ * FF_, C_,  FF_);
  }
  repack_t_kernel<<<(V_ * C_) / 256, 256, 0, stream>>>(Wlm, wlmT, V_, C_);

  embed_kernel<<<M_, 128, 0, stream>>>(idx, labs, tok, pos, labe, X);

  dim3 gblk(128);
  for (int l = 0; l < L_; ++l) {
    ln_kernel<<<M_, 32, 0, stream>>>(X, ln1g + l * C_, ln1b + l * C_, Hb);
    // fused Q|K projection -> QKb[M, 1024] bf16 (cols 0..511 = q, 512..1023 = k; col = h*HS+d)
    gemm_bf16_kernel<false, false, false, true><<<dim3(2 * C_ / 64, M_ / 256), gblk, 0, stream>>>(
        Hb, wqkvT + (size_t)l * 1536 * C_, nullptr, nullptr, QKb, C_, 2 * C_);
    // V projection, stored transposed as vT[b][h][d][s] bf16
    gemm_bf16_kernel<false, false, true, true><<<dim3(C_ / 64, M_ / 256), gblk, 0, stream>>>(
        Hb, wqkvT + (size_t)l * 1536 * C_ + (size_t)2 * C_ * C_, nullptr, nullptr, VTb, C_, 0);
    attn_kernel<<<dim3(T_ / 16, B_ * H_), dim3(32), 0, stream>>>(QKb, VTb, Ob);
    // x = x + o @ Wo + bo
    gemm_bf16_kernel<false, true, false, false><<<dim3(C_ / 64, M_ / 256), gblk, 0, stream>>>(
        Ob, woT + (size_t)l * C_ * C_, bo + l * C_, X, X, C_, C_);
    ln_kernel<<<M_, 32, 0, stream>>>(X, ln2g + l * C_, ln2b + l * C_, Hb);
    // ff = relu(h @ W1 + b1), bf16
    gemm_bf16_kernel<true, false, false, true><<<dim3(FF_ / 64, M_ / 256), gblk, 0, stream>>>(
        Hb, w1T + (size_t)l * FF_ * C_, b1 + l * FF_, nullptr, FFb, C_, FF_);
    // x = x + ff @ W2 + b2
    gemm_bf16_kernel<false, true, false, false><<<dim3(C_ / 64, M_ / 256), gblk, 0, stream>>>(
        FFb, w2T + (size_t)l * C_ * FF_, b2 + l * C_, X, X, FF_, C_);
  }
  ln_kernel<<<M_, 32, 0, stream>>>(X, lnfg, lnfb, Hb);

  float* logits = (float*)d_out;
  gemm_bf16_kernel<false, false, false, false><<<dim3(V_ / 64, M_ / 256), gblk, 0, stream>>>(
      Hb, wlmT, blm, nullptr, logits, C_, V_);

  loss_rows_kernel<<<M_, 256, 0, stream>>>(logits, tgt, terms);
  loss_reduce_kernel<<<1, 256, 0, stream>>>(terms, logits + (size_t)M_ * V_);
}
